// GraphGNNModel_22265110462802
// MI455X (gfx1250) — compile-verified, matrix operand order checked by reading, not run on hardware
//
#include <hip/hip_runtime.h>

// ---------------------------------------------------------------------------
// CDNA5 (gfx1250) fused NNConv GNN.
// wave32; matrix work on v_wmma_f32_16x16x32_bf16 (f32 accumulate).
// ---------------------------------------------------------------------------

typedef __attribute__((ext_vector_type(16))) __bf16 v16bf;
typedef __attribute__((ext_vector_type(8)))  float  v8f;

#define WMMA_BF16(a, b, c) \
  __builtin_amdgcn_wmma_f32_16x16x32_bf16(false, (a), false, (b), (short)0, (c), false, false)

// ---------------------------------------------------------------------------
// Weight pre-pack: row-major f32 [K, Ntot] -> bf16 WMMA B-fragments.
// Fragment layout (16-bit B matrix, 32x16 K-major tile, wave32):
//   lane = {n = lane&15, kbase = (lane>>4)*8}
//   elem e<8  : k = kbase + e          (VGPRs 0..3, 2 packed per VGPR)
//   elem e>=8 : k = 16 + kbase + (e-8) (VGPRs 4..7)
// Packed index: ((kc*NT + nt)*32 + lane)*16 + e   -> per-lane contiguous 32B.
// ---------------------------------------------------------------------------
__global__ void pack_b_frags(const float* __restrict__ W, __bf16* __restrict__ out,
                             int K, int Ntot) {
  int NT = Ntot >> 4;
  int KC = (K + 31) >> 5;
  int total = KC * NT * 512;
  for (int idx = blockIdx.x * blockDim.x + threadIdx.x; idx < total;
       idx += gridDim.x * blockDim.x) {
    int e    = idx & 15;
    int lane = (idx >> 4) & 31;
    int t    = idx >> 9;            // kc*NT + nt
    int nt = t % NT, kc = t / NT;
    int n = lane & 15;
    int kbase = (lane >> 4) * 8;
    int kl = (e < 8) ? (kbase + e) : (16 + kbase + (e - 8));
    int kg = kc * 32 + kl;
    int ng = nt * 16 + n;
    float v = (kg < K) ? W[(size_t)kg * Ntot + ng] : 0.0f;
    out[idx] = (__bf16)v;
  }
}

__global__ void zero_f32(float* __restrict__ p, size_t n) {
  for (size_t i = blockIdx.x * (size_t)blockDim.x + threadIdx.x; i < n;
       i += (size_t)gridDim.x * blockDim.x)
    p[i] = 0.0f;
}

// A-fragment (16x32 bf16, M-major) from LDS row-major [16, stride] at K-chunk kc.
__device__ inline v16bf load_A_lds(const __bf16* s, int stride, int kc, int lane) {
  int m = lane & 15;
  int kbase = ((lane >> 4) * 8) + kc * 32;
  v16bf a;
#pragma unroll
  for (int e = 0; e < 8; ++e) {
    a[e]     = s[m * stride + kbase + e];
    a[e + 8] = s[m * stride + kbase + 16 + e];
  }
  return a;
}

__device__ inline v16bf load_B_packed(const __bf16* __restrict__ packed, int tileIdx,
                                      int lane) {
  // 32 contiguous bytes per lane -> two global_load_b128
  return *(const v16bf*)(packed + ((size_t)(tileIdx * 32 + lane) << 4));
}

// ---------------------------------------------------------------------------
// Fused NNConv edge kernel. One wave handles 16 edges:
//   H  = relu(edge_attr[16,8] @ eW1[8,64] + eb1)            (4 WMMAs, K padded)
//   W  = relu(H @ eW2[64, CIN*32] + eb2)  tile-by-tile      (2 WMMAs / tile)
//   msg[e,o] = sum_i x[src[e],i] * W[e,i,o]   (folded per tile: i fixed/tile)
//   atomicAdd(agg[dst[e], o], msg[e,o])
// ---------------------------------------------------------------------------
template <int CIN>
__global__ __launch_bounds__(256) void nnconv_edge(
    const float* __restrict__ x,           // [N, CIN] node features (gathered)
    const int* __restrict__ edge_index,    // [2, E]
    const float* __restrict__ edge_attr,   // [E, 8]
    const __bf16* __restrict__ W1f,        // GEMM1 B-frags (KC=1, NT=4)
    const float* __restrict__ eb1,         // [64]
    const __bf16* __restrict__ W2f,        // GEMM2 B-frags (KC=2, NT=CIN*2)
    const float* __restrict__ eb2,         // [CIN*32]
    float* __restrict__ agg,               // [N, 32] (pre-zeroed)
    int E_) {
  constexpr int COUT = 32;
  constexpr int NT2 = (CIN * COUT) / 16;   // 32 (layer1) or 64 (layer2)

  const int lane = threadIdx.x & 31;
  const int wave = threadIdx.x >> 5;
  const int nTiles = (E_ + 15) >> 4;
  const int tile = blockIdx.x * 8 + wave;
  if (tile >= nTiles) return;              // wave-uniform; EXEC stays full
  const int e0 = tile << 4;

  __shared__ __bf16 sH[8][16 * 64];
  __shared__ float  sX[8][16 * CIN];
  __shared__ int    sDst[8][16];
  __bf16* myH = sH[wave];
  float*  myX = sX[wave];
  int*    myDst = sDst[wave];

  // ---- stage dst indices + gathered x[src] rows (clamped for tail tile) ----
  if (lane < 16) {
    int eg = e0 + lane;
    myDst[lane] = edge_index[E_ + (eg < E_ ? eg : E_ - 1)];
  }
  for (int idx = lane; idx < 16 * CIN; idx += 32) {
    int m = idx / CIN, c = idx % CIN;
    int eg = e0 + m;
    int srcv = edge_index[eg < E_ ? eg : E_ - 1];
    myX[idx] = x[(size_t)srcv * CIN + c];
  }

  // ---- A-fragment for edge_attr: K=8 zero-padded to 32 --------------------
  // lanes 0-15 cover K=0..7 (elems 0..7); everything else is the zero pad.
  v16bf aE;
#pragma unroll
  for (int e = 0; e < 16; ++e) aE[e] = (__bf16)0.0f;
  {
    int m = lane & 15;
    int eg = e0 + m;
    if (lane < 16 && eg < E_) {
      const float* row = edge_attr + (size_t)eg * 8;
#pragma unroll
      for (int e = 0; e < 8; ++e) aE[e] = (__bf16)row[e];
    }
  }

  const int n = lane & 15;
  const int half = lane >> 4;

  // ---- GEMM1: [16x32(pad)] @ [32x64] -> H, relu+bias, stash bf16 in LDS ---
#pragma unroll
  for (int nt = 0; nt < 4; ++nt) {
    v8f c = {};
    v16bf b = load_B_packed(W1f, nt, lane);
    c = WMMA_BF16(aE, b, c);
    int col = nt * 16 + n;
    float bias = eb1[col];
#pragma unroll
    for (int r = 0; r < 8; ++r) {
      int m = half * 8 + r;                          // C/D layout: M = r + half*8
      myH[m * 64 + col] = (__bf16)fmaxf(c[r] + bias, 0.0f);
    }
  }

  // ---- GEMM2 (K=64, 2 chunks) fused with per-edge contraction -------------
  v16bf a0 = load_A_lds(myH, 64, 0, lane);
  v16bf a1 = load_A_lds(myH, 64, 1, lane);

  v8f accL = {};   // msg columns o = n
  v8f accH = {};   // msg columns o = 16 + n
  for (int t = 0; t < NT2; ++t) {
    v8f c = {};
    v16bf b0 = load_B_packed(W2f, t, lane);          // K chunk 0
    v16bf b1 = load_B_packed(W2f, NT2 + t, lane);    // K chunk 1
    c = WMMA_BF16(a0, b0, c);
    c = WMMA_BF16(a1, b1, c);

    int n0 = t * 16;
    int i = n0 / COUT;                 // COUT=32 >= 16: one input chan / tile
    float bias = eb2[n0 + n];
    if (t & 1) {
#pragma unroll
      for (int r = 0; r < 8; ++r) {
        int m = half * 8 + r;
        float w = fmaxf(c[r] + bias, 0.0f);
        accH[r] += myX[m * CIN + i] * w;
      }
    } else {
#pragma unroll
      for (int r = 0; r < 8; ++r) {
        int m = half * 8 + r;
        float w = fmaxf(c[r] + bias, 0.0f);
        accL[r] += myX[m * CIN + i] * w;
      }
    }
  }

  // ---- scatter-add into destination nodes ---------------------------------
#pragma unroll
  for (int r = 0; r < 8; ++r) {
    int m = half * 8 + r;
    if (e0 + m < E_) {
      size_t base = (size_t)myDst[m] * 32;
      atomicAdd(&agg[base + n], accL[r]);
      atomicAdd(&agg[base + 16 + n], accH[r]);
    }
  }
}

// h[v,o] = act(agg[v,o] + sum_i xin[v,i]*root[i,o] + bias[o])
template <int CIN, bool RELU>
__global__ void node_update(const float* __restrict__ agg, const float* __restrict__ xin,
                            const float* __restrict__ root, const float* __restrict__ bias,
                            float* __restrict__ out, int Nn) {
  int idx = blockIdx.x * blockDim.x + threadIdx.x;
  if (idx >= Nn * 32) return;
  int v = idx >> 5, o = idx & 31;
  float s = agg[idx] + bias[o];
  const float* xv = xin + (size_t)v * CIN;
#pragma unroll
  for (int i = 0; i < CIN; ++i) s += xv[i] * root[i * 32 + o];
  out[idx] = RELU ? fmaxf(s, 0.0f) : s;
}

// layer-2 node update + global mean-pool accumulation
__global__ void node_update_pool(const float* __restrict__ agg, const float* __restrict__ h1,
                                 const float* __restrict__ root, const float* __restrict__ bias,
                                 const int* __restrict__ batch_idx,
                                 float* __restrict__ pool_sum, float* __restrict__ cnt,
                                 int Nn) {
  int idx = blockIdx.x * blockDim.x + threadIdx.x;
  if (idx >= Nn * 32) return;
  int v = idx >> 5, o = idx & 31;
  float s = agg[idx] + bias[o];
  const float* hv = h1 + (size_t)v * 32;
#pragma unroll
  for (int i = 0; i < 32; ++i) s += hv[i] * root[i * 32 + o];
  int g = batch_idx[v];
  atomicAdd(&pool_sum[(size_t)g * 32 + o], s);
  if (o == 0) atomicAdd(&cnt[g], 1.0f);
}

__device__ inline float mlp32_step(float z, float* sbuf, int lane,
                                   const float* __restrict__ W,
                                   const float* __restrict__ b) {
  sbuf[lane] = z;                         // same-wave LDS: in-order, no barrier
  float s = b[lane];
#pragma unroll
  for (int k = 0; k < 32; ++k) s += sbuf[k] * W[k * 32 + lane];
  return fmaxf(s, 0.0f);
}

// per-graph head: mean, 3x Linear(32,32)+ReLU, Linear(32,1). One wave / graph.
__global__ __launch_bounds__(256) void graph_head(
    const float* __restrict__ pool_sum, const float* __restrict__ cnt,
    const float* __restrict__ nW1, const float* __restrict__ nb1,
    const float* __restrict__ nW2, const float* __restrict__ nb2,
    const float* __restrict__ nW3, const float* __restrict__ nb3,
    const float* __restrict__ hW, const float* __restrict__ hb,
    float* __restrict__ out, int Gn) {
  int lane = threadIdx.x & 31;
  int wave = threadIdx.x >> 5;
  int g = blockIdx.x * 8 + wave;
  if (g >= Gn) return;
  __shared__ float sz[8][32];
  float c = fmaxf(cnt[g], 1.0f);
  float z = pool_sum[(size_t)g * 32 + lane] / c;
  z = mlp32_step(z, sz[wave], lane, nW1, nb1);
  z = mlp32_step(z, sz[wave], lane, nW2, nb2);
  z = mlp32_step(z, sz[wave], lane, nW3, nb3);
  sz[wave][lane] = z;
  if (lane == 0) {
    float s = hb[0];
#pragma unroll
    for (int k = 0; k < 32; ++k) s += sz[wave][k] * hW[k];
    out[g] = s;
  }
}

// ---------------------------------------------------------------------------
extern "C" void kernel_launch(void* const* d_in, const int* in_sizes, int n_in,
                              void* d_out, int out_size, void* d_ws, size_t ws_size,
                              hipStream_t stream) {
  const float* x          = (const float*)d_in[0];
  const int*   edge_index = (const int*)d_in[1];
  const float* edge_attr  = (const float*)d_in[2];
  const int*   batch_idx  = (const int*)d_in[3];
  const float* eW1_1 = (const float*)d_in[4];
  const float* eb1_1 = (const float*)d_in[5];
  const float* eW2_1 = (const float*)d_in[6];
  const float* eb2_1 = (const float*)d_in[7];
  const float* root1 = (const float*)d_in[8];
  const float* bias1 = (const float*)d_in[9];
  const float* eW1_2 = (const float*)d_in[10];
  const float* eb1_2 = (const float*)d_in[11];
  const float* eW2_2 = (const float*)d_in[12];
  const float* eb2_2 = (const float*)d_in[13];
  const float* root2 = (const float*)d_in[14];
  const float* bias2 = (const float*)d_in[15];
  const float* nW1 = (const float*)d_in[16];
  const float* nb1 = (const float*)d_in[17];
  const float* nW2 = (const float*)d_in[18];
  const float* nb2 = (const float*)d_in[19];
  const float* nW3 = (const float*)d_in[20];
  const float* nb3 = (const float*)d_in[21];
  const float* hW  = (const float*)d_in[22];
  const float* hb  = (const float*)d_in[23];

  const int N = in_sizes[0] / 16;
  const int E = in_sizes[1] / 2;
  const int G = out_size;          // C_OUT == 1
  float* out = (float*)d_out;

  // ---- workspace carve-up (256B aligned) ----------------------------------
  char* ws = (char*)d_ws;
  size_t off = 0;
  auto carve = [&](size_t bytes) {
    size_t o = off;
    off = (off + bytes + 255) & ~(size_t)255;
    return o;
  };
  float* agg1     = (float*)(ws + carve((size_t)N * 32 * 4));
  float* h1       = (float*)(ws + carve((size_t)N * 32 * 4));
  float* agg2     = (float*)(ws + carve((size_t)N * 32 * 4));
  float* pool_sum = (float*)(ws + carve((size_t)G * 32 * 4));
  float* cnt      = (float*)(ws + carve((size_t)G * 4));
  size_t zero_bytes = off;                       // zero everything up to here
  __bf16* W1f_1 = (__bf16*)(ws + carve(1 * 4 * 512 * 2));    // eW1_1 (K pad 32)
  __bf16* W2f_1 = (__bf16*)(ws + carve(2 * 32 * 512 * 2));   // eW2_1
  __bf16* W1f_2 = (__bf16*)(ws + carve(1 * 4 * 512 * 2));    // eW1_2
  __bf16* W2f_2 = (__bf16*)(ws + carve(2 * 64 * 512 * 2));   // eW2_2
  (void)ws_size;

  // ---- 0) zero accumulators ----------------------------------------------
  zero_f32<<<1024, 256, 0, stream>>>((float*)ws, zero_bytes / 4);

  // ---- 1) pack bf16 B-fragments (tiny; cached in L2 thereafter) ----------
  pack_b_frags<<<8, 256, 0, stream>>>(eW1_1, W1f_1, 8, 64);
  pack_b_frags<<<128, 256, 0, stream>>>(eW2_1, W2f_1, 64, 512);
  pack_b_frags<<<8, 256, 0, stream>>>(eW1_2, W1f_2, 8, 64);
  pack_b_frags<<<256, 256, 0, stream>>>(eW2_2, W2f_2, 64, 1024);

  const int nTiles = (E + 15) / 16;
  const int edgeBlocks = (nTiles + 7) / 8;
  const int nodeBlocks = (N * 32 + 255) / 256;

  // ---- 2) layer 1: fused edge MLP + contraction + scatter ----------------
  nnconv_edge<16><<<edgeBlocks, 256, 0, stream>>>(x, edge_index, edge_attr,
                                                  W1f_1, eb1_1, W2f_1, eb2_1,
                                                  agg1, E);
  // ---- 3) h1 = relu(agg1 + x@root1 + bias1) ------------------------------
  node_update<16, true><<<nodeBlocks, 256, 0, stream>>>(agg1, x, root1, bias1, h1, N);

  // ---- 4) layer 2 ---------------------------------------------------------
  nnconv_edge<32><<<edgeBlocks, 256, 0, stream>>>(h1, edge_index, edge_attr,
                                                  W1f_2, eb1_2, W2f_2, eb2_2,
                                                  agg2, E);
  // ---- 5) h2 = agg2 + h1@root2 + bias2, pooled into graph sums -----------
  node_update_pool<<<nodeBlocks, 256, 0, stream>>>(agg2, h1, root2, bias2,
                                                   batch_idx, pool_sum, cnt, N);

  // ---- 6) per-graph mean + MLP + head ------------------------------------
  graph_head<<<(G + 7) / 8, 256, 0, stream>>>(pool_sum, cnt, nW1, nb1, nW2, nb2,
                                              nW3, nb3, hW, hb, out, G);
}
